// DependencyLSTMLocalModel_81037442941334
// MI455X (gfx1250) — compile-verified, matrix operand order checked by reading, not run on hardware
//
#include <hip/hip_runtime.h>
#include <hip/hip_bf16.h>
#include <math.h>

// ---------------------------------------------------------------------------
// DependencyLSTM on gfx1250 (MI455X, wave32):
//   embed+dep-mean -> batched WMMA-bf16 input GEMM (TDM-staged B operands)
//   -> register-resident WMMA recurrence (Whh hoisted to VGPRs, fused max-pool)
//   -> tiny classifier.
// ---------------------------------------------------------------------------

typedef __attribute__((ext_vector_type(16))) __bf16 v16bf;
typedef __attribute__((ext_vector_type(8)))  float  v8f;
typedef __attribute__((ext_vector_type(4)))  unsigned int u32x4;
typedef __attribute__((ext_vector_type(8)))  int  i32x8;
typedef __attribute__((ext_vector_type(4)))  int  i32x4;

#define B_   32
#define S_   512
#define E_   300
#define EP_  320      // E padded to multiple of 32
#define H_   128
#define G4_  512      // 4*H
#define D_   8
#define KC_IN  10     // EP/32
#define KC_REC 4      // H/32

#if defined(__has_builtin)
#if __has_builtin(__builtin_amdgcn_tensor_load_to_lds) && \
    __has_builtin(__builtin_amdgcn_s_wait_tensorcnt)
#define HAVE_TDM 1
#endif
#endif

union FragU { uint4 q[2]; v16bf v; };

// A-fragment (16x32 bf16, per-lane): VGPR0..3 hold k = base+8*half+{0..7},
// VGPR4..7 hold k = base+16+8*half+{0..7}  -> two contiguous 16B loads.
__device__ __forceinline__ v16bf load_afrag(const __bf16* row, int kbase, int half) {
    FragU f;
    f.q[0] = *reinterpret_cast<const uint4*>(row + kbase + 8 * half);
    f.q[1] = *reinterpret_cast<const uint4*>(row + kbase + 16 + 8 * half);
    return f.v;
}

__device__ __forceinline__ v8f wmma_bf16(v16bf a, v16bf b, v8f c) {
    return __builtin_amdgcn_wmma_f32_16x16x32_bf16(
        /*neg_a=*/false, a, /*neg_b=*/false, b,
        /*c_mod=*/(short)0, c, /*reuse_a=*/false, /*reuse_b=*/false);
}

__device__ __forceinline__ float sigm(float x) { return 1.0f / (1.0f + __expf(-x)); }

#ifdef HAVE_TDM
// TDM: contiguous 32KB (8192 dwords) global -> LDS copy, one descriptor.
// D# group0: count=1 | lds_addr | global_addr[56:0] | type=2.
// D# group1: data_size=4B; tensor_dim0=tile_dim0=stride=8192; dims1/2 = 1.
__device__ __forceinline__ void tdm_load_slab32k(const void* gsrc, unsigned ldsOff) {
    unsigned long long ga = (unsigned long long)(uintptr_t)gsrc;
    u32x4 g0 = { 1u,                                  // count=1 (valid descriptor)
                 ldsOff,                              // lds byte address
                 (unsigned)(ga & 0xFFFFFFFFu),        // global addr [31:0]
                 (unsigned)(((ga >> 32) & 0x01FFFFFFu) | 0x80000000u) }; // [56:32] | type=2
    i32x8 g1 = { 0x00020000,    // data_size = 2 (4 bytes)
                 0x20000000,    // tensor_dim0[15:0]=8192 << 16
                 0x00010000,    // tensor_dim0[31:16]=0 | tensor_dim1[15:0]=1
                 0x20000000,    // tensor_dim1[31:16]=0 | tile_dim0=8192
                 1,             // tile_dim1=1, tile_dim2=0
                 8192,          // tensor_dim0_stride low32
                 0, 0 };
    i32x4 z4 = { 0, 0, 0, 0 };
#if __clang_major__ >= 23
    i32x8 z8 = { 0, 0, 0, 0, 0, 0, 0, 0 };
    __builtin_amdgcn_tensor_load_to_lds(g0, g1, z4, z4, z8, 0);
#else
    __builtin_amdgcn_tensor_load_to_lds(g0, g1, z4, z4, 0);
#endif
}
#endif

// ---------------------------------------------------------------------------
// K0: swizzle weights (f32 row-major [4H, K]) into per-lane WMMA B-fragment
// layout dst[dir][kChunk][nTile][lane][16 bf16]  (32 contiguous bytes/lane;
// each (dir,kChunk) slab is 32KB contiguous -> one TDM transfer).
// Lane element v16 -> k_local = 16*(v>>2) + 8*(lane>>4) + 2*(v&3) + p.
// ---------------------------------------------------------------------------
__global__ void swizzle_weights(const float* __restrict__ Wf,
                                const float* __restrict__ Wb,
                                __bf16* __restrict__ dst,
                                int kChunks, int Ksrc) {
    int total = 2 * kChunks * 32 * 512;
    for (int idx = blockIdx.x * blockDim.x + threadIdx.x; idx < total;
         idx += gridDim.x * blockDim.x) {
        int perDir = kChunks * 32 * 512;
        int dir  = idx / perDir;
        int rem  = idx % perDir;
        int kc   = rem / (32 * 512);
        int rem2 = rem % (32 * 512);
        int nt   = rem2 / 512;
        int rem3 = rem2 % 512;
        int lane = rem3 / 16;
        int v16  = rem3 % 16;
        int v = v16 >> 1, p = v16 & 1, half = lane >> 4, nl = lane & 15;
        int kl = 16 * (v >> 2) + 8 * half + 2 * (v & 3) + p;
        int k  = kc * 32 + kl;
        int n  = nt * 16 + nl;
        const float* src = dir ? Wb : Wf;
        float val = (k < Ksrc) ? src[n * Ksrc + k] : 0.0f;
        dst[idx] = (__bf16)val;
    }
}

// ---------------------------------------------------------------------------
// K1: embedding gather + masked dep-mean -> x_bf16 [B*S, EP] (zero-padded)
// ---------------------------------------------------------------------------
__global__ void embed_kernel(const int* __restrict__ word_ids,
                             const int* __restrict__ deps_ids,
                             const float* __restrict__ wtab,
                             const float* __restrict__ dtab,
                             __bf16* __restrict__ x) {
    int bs = blockIdx.x;             // b*S + s
    int b  = bs >> 9;
    int s  = bs & 511;
    int w  = word_ids[(b * 3 + 1) * S_ + s];
    const int* dep = deps_ids + bs * D_;
    int ids[D_]; bool vld[D_]; float cnt = 0.0f;
#pragma unroll
    for (int d = 0; d < D_; ++d) {
        int id = dep[d];
        ids[d] = id;
        vld[d] = (id != 0) && (id != 1);
        cnt += vld[d] ? 1.0f : 0.0f;
    }
    float inv = (cnt > 0.0f) ? (0.5f / cnt) : 0.0f;
    for (int e = threadIdx.x; e < EP_; e += blockDim.x) {
        float out = 0.0f;
        if (e < E_) {
            float wv = wtab[(size_t)w * E_ + e];
            float sum = 0.0f;
#pragma unroll
            for (int d = 0; d < D_; ++d)
                if (vld[d]) sum += dtab[ids[d] * E_ + e];
            out = (cnt > 0.0f) ? (0.5f * wv + inv * sum) : wv;
        }
        x[(size_t)bs * EP_ + e] = (__bf16)out;
    }
}

// ---------------------------------------------------------------------------
// K2: batched input projection G[dir][s][b][4H] = x @ Wih^T + bias.
// Block = 8 waves; one (dir, mTile of 16 rows); wave covers 4 nTiles.
// B operands: per-kChunk 32KB slab staged into LDS by TDM (double-buffered,
// wave0 issues + waits TENSORcnt, barriers publish to other waves).
// ---------------------------------------------------------------------------
__global__ void input_gemm(const __bf16* __restrict__ x,
                           const __bf16* __restrict__ wih_sw,
                           const float* __restrict__ bias_f,
                           const float* __restrict__ bias_b,
                           float* __restrict__ G) {
    int blk   = blockIdx.x;
    int dir   = blk >> 10;
    int mTile = blk & 1023;
    int w    = threadIdx.x >> 5;
    int lane = threadIdx.x & 31;
    int half = lane >> 4;
    int nl   = lane & 15;
    const float* bias = dir ? bias_b : bias_f;

    v8f acc[4];
#pragma unroll
    for (int j = 0; j < 4; ++j) {
        float bv = bias[(w * 4 + j) * 16 + nl];
#pragma unroll
        for (int r = 0; r < 8; ++r) acc[j][r] = bv;
    }

    const __bf16* arow = x + (size_t)(mTile * 16 + nl) * EP_;

#ifdef HAVE_TDM
    __shared__ __bf16 bstage[2][32 * 32 * 16];   // 2 x 32KB double buffer
    bool isW0 = (threadIdx.x < 32);
    if (isW0)
        tdm_load_slab32k(wih_sw + (size_t)(dir * KC_IN + 0) * (32 * 32 * 16),
                         (unsigned)(uintptr_t)&bstage[0][0]);
#pragma unroll 1
    for (int kc = 0; kc < KC_IN; ++kc) {
        if (isW0) {
            if (kc + 1 < KC_IN) {
                tdm_load_slab32k(wih_sw + (size_t)(dir * KC_IN + kc + 1) * (32 * 32 * 16),
                                 (unsigned)(uintptr_t)&bstage[(kc + 1) & 1][0]);
                __builtin_amdgcn_s_wait_tensorcnt(1);   // slab kc complete
            } else {
                __builtin_amdgcn_s_wait_tensorcnt(0);
            }
        }
        __syncthreads();                                // publish slab kc
        v16bf af = load_afrag(arow, kc * 32, half);
        const __bf16* slab = &bstage[kc & 1][0];
#pragma unroll
        for (int j = 0; j < 4; ++j) {
            int nt = w * 4 + j;
            const uint4* bp = reinterpret_cast<const uint4*>(slab + (nt * 32 + lane) * 16);
            FragU bfr; bfr.q[0] = bp[0]; bfr.q[1] = bp[1];
            acc[j] = wmma_bf16(af, bfr.v, acc[j]);
        }
        __syncthreads();                                // reads done; buffer reusable
    }
#else
#pragma unroll 2
    for (int kc = 0; kc < KC_IN; ++kc) {
        v16bf af = load_afrag(arow, kc * 32, half);
#pragma unroll
        for (int j = 0; j < 4; ++j) {
            int nt = w * 4 + j;
            const uint4* bp = reinterpret_cast<const uint4*>(
                wih_sw + (size_t)(((dir * KC_IN + kc) * 32 + nt) * 32 + lane) * 16);
            FragU bfr; bfr.q[0] = bp[0]; bfr.q[1] = bp[1];
            acc[j] = wmma_bf16(af, bfr.v, acc[j]);
        }
    }
#endif

    int b  = (mTile * 16) >> 9;       // tile never crosses a batch boundary (512%16==0)
    int s0 = (mTile * 16) & 511;
#pragma unroll
    for (int j = 0; j < 4; ++j) {
        int n = (w * 4 + j) * 16 + nl;
#pragma unroll
        for (int r = 0; r < 8; ++r) {
            int s = s0 + r + 8 * half;
            G[(((size_t)dir * S_ + s) * B_ + b) * G4_ + n] = acc[j][r];
        }
    }
}

// ---------------------------------------------------------------------------
// K3: serial recurrence. 1 block per direction, 8 waves.
// Wave w owns hidden slice j in [16w, 16w+16): computes i/f/g/o tiles for both
// batch tiles -> lane-local LSTM update; Whh fragments HOISTED into registers
// for the whole sequence; c + running max in registers; h (bf16, 8KB) via LDS.
// ---------------------------------------------------------------------------
__global__ void lstm_rec(const float* __restrict__ G,
                         const __bf16* __restrict__ whh_sw,
                         float* __restrict__ pooled) {
    int dir  = blockIdx.x;
    int w    = threadIdx.x >> 5;
    int lane = threadIdx.x & 31;
    int half = lane >> 4;
    int nl   = lane & 15;

    __shared__ __bf16 h_lds[B_][H_];
    for (int i = threadIdx.x; i < B_ * H_; i += blockDim.x)
        (&h_lds[0][0])[i] = (__bf16)0.0f;

    // Hoist this wave's 16 Whh B-fragments (4 gates x 4 kChunks) into VGPRs.
    FragU whh[4][KC_REC];
#pragma unroll
    for (int g = 0; g < 4; ++g)
#pragma unroll
        for (int kc = 0; kc < KC_REC; ++kc) {
            int nt = g * 8 + w;
            const uint4* bp = reinterpret_cast<const uint4*>(
                whh_sw + (size_t)(((dir * KC_REC + kc) * 32 + nt) * 32 + lane) * 16);
            whh[g][kc].q[0] = bp[0];
            whh[g][kc].q[1] = bp[1];
        }

    float cst[2][8];
    float pmax[2][8];
#pragma unroll
    for (int mt = 0; mt < 2; ++mt)
#pragma unroll
        for (int r = 0; r < 8; ++r) { cst[mt][r] = 0.0f; pmax[mt][r] = -1e30f; }

    __syncthreads();

#pragma unroll 1
    for (int step = 0; step < S_; ++step) {
        int s_x = dir ? (S_ - 1 - step) : step;
        const float* Gstep = G + ((size_t)dir * S_ + s_x) * B_ * G4_;

        // init accumulators from precomputed input gates
        v8f acc[2][4];
#pragma unroll
        for (int mt = 0; mt < 2; ++mt)
#pragma unroll
            for (int g = 0; g < 4; ++g) {
                int n = g * H_ + w * 16 + nl;
#pragma unroll
                for (int r = 0; r < 8; ++r) {
                    int b = mt * 16 + r + 8 * half;
                    acc[mt][g][r] = Gstep[b * G4_ + n];
                }
            }

        // gates += h_prev @ Whh^T  (WMMA over K=128, weights in registers)
#pragma unroll
        for (int kc = 0; kc < KC_REC; ++kc) {
            v16bf af[2];
#pragma unroll
            for (int mt = 0; mt < 2; ++mt)
                af[mt] = load_afrag(&h_lds[mt * 16 + nl][0], kc * 32, half);
#pragma unroll
            for (int g = 0; g < 4; ++g)
#pragma unroll
                for (int mt = 0; mt < 2; ++mt)
                    acc[mt][g] = wmma_bf16(af[mt], whh[g][kc].v, acc[mt][g]);
        }

        // prefetch next step's input gates into cache (global_prefetch_b8)
        if (step + 1 < S_) {
            int s_n = dir ? (S_ - 2 - step) : (step + 1);
            const float* Gnext = G + ((size_t)dir * S_ + s_n) * B_ * G4_;
            __builtin_prefetch(Gnext + threadIdx.x * 64, 0, 1);
        }

        __syncthreads();   // all h_lds reads complete before overwrite

        // lane-local LSTM update + fused max-pool
#pragma unroll
        for (int mt = 0; mt < 2; ++mt)
#pragma unroll
            for (int r = 0; r < 8; ++r) {
                float ig = sigm(acc[mt][0][r]);
                float fg = sigm(acc[mt][1][r]);
                float gg = tanhf(acc[mt][2][r]);
                float og = sigm(acc[mt][3][r]);
                float cn = fg * cst[mt][r] + ig * gg;
                cst[mt][r] = cn;
                float hn = og * tanhf(cn);
                pmax[mt][r] = fmaxf(pmax[mt][r], hn);
                int b = mt * 16 + r + 8 * half;
                int j = w * 16 + nl;
                h_lds[b][j] = (__bf16)hn;
            }

        __syncthreads();   // h writes visible before next step's reads
    }

    // pooled[dir][b][j]
#pragma unroll
    for (int mt = 0; mt < 2; ++mt)
#pragma unroll
        for (int r = 0; r < 8; ++r) {
            int b = mt * 16 + r + 8 * half;
            int j = w * 16 + nl;
            pooled[((size_t)dir * B_ + b) * H_ + j] = pmax[mt][r];
        }
}

// ---------------------------------------------------------------------------
// K4: logits = pooled_cat @ W_cls^T + b_cls   (32x5, trivial)
// ---------------------------------------------------------------------------
__global__ void classify(const float* __restrict__ pooled,
                         const float* __restrict__ Wc,
                         const float* __restrict__ bc,
                         float* __restrict__ out) {
    int idx = threadIdx.x;
    if (idx >= B_ * 5) return;
    int b = idx / 5, t = idx % 5;
    float s = bc[t];
#pragma unroll 4
    for (int k = 0; k < 2 * H_; ++k) {
        float feat = (k < H_) ? pooled[b * H_ + k]
                              : pooled[(B_ + b) * H_ + (k - H_)];
        s += feat * Wc[t * (2 * H_) + k];
    }
    out[b * 5 + t] = s;
}

// ---------------------------------------------------------------------------
extern "C" void kernel_launch(void* const* d_in, const int* in_sizes, int n_in,
                              void* d_out, int out_size, void* d_ws, size_t ws_size,
                              hipStream_t stream) {
    (void)in_sizes; (void)n_in; (void)out_size; (void)ws_size;
    const int*   word_ids = (const int*)  d_in[0];
    const int*   deps_ids = (const int*)  d_in[1];
    const float* wtab     = (const float*)d_in[2];
    const float* dtab     = (const float*)d_in[3];
    const float* Wih_f    = (const float*)d_in[4];
    const float* Whh_f    = (const float*)d_in[5];
    const float* b_f      = (const float*)d_in[6];
    const float* Wih_b    = (const float*)d_in[7];
    const float* Whh_b    = (const float*)d_in[8];
    const float* b_b      = (const float*)d_in[9];
    const float* W_cls    = (const float*)d_in[10];
    const float* b_cls    = (const float*)d_in[11];
    float* out = (float*)d_out;

    char* ws = (char*)d_ws;
    size_t offX    = 0;                                           // x bf16 [16384][320]
    size_t offWih  = offX   + (size_t)B_ * S_ * EP_ * 2;          // +10,485,760
    size_t offWhh  = offWih + (size_t)2 * KC_IN  * 32 * 512 * 2;  // +655,360
    size_t offG    = offWhh + (size_t)2 * KC_REC * 32 * 512 * 2;  // +262,144
    size_t offPool = offG   + (size_t)2 * S_ * B_ * G4_ * 4;      // +67,108,864
    __bf16* x_bf   = (__bf16*)(ws + offX);
    __bf16* wih_sw = (__bf16*)(ws + offWih);
    __bf16* whh_sw = (__bf16*)(ws + offWhh);
    float*  G      = (float*) (ws + offG);
    float*  pooled = (float*) (ws + offPool);

    {   // K0: weight swizzles
        int nIh = 2 * KC_IN * 32 * 512;
        swizzle_weights<<<(nIh + 255) / 256, 256, 0, stream>>>(Wih_f, Wih_b, wih_sw, KC_IN, E_);
        int nHh = 2 * KC_REC * 32 * 512;
        swizzle_weights<<<(nHh + 255) / 256, 256, 0, stream>>>(Whh_f, Whh_b, whh_sw, KC_REC, H_);
    }
    // K1: embeddings
    embed_kernel<<<B_ * S_, 128, 0, stream>>>(word_ids, deps_ids, wtab, dtab, x_bf);
    // K2: batched input GEMM (both directions), TDM-staged B operands
    input_gemm<<<2 * (B_ * S_ / 16), 256, 0, stream>>>(x_bf, wih_sw, b_f, b_b, G);
    // K3: recurrence + fused max-pool
    lstm_rec<<<2, 256, 0, stream>>>(G, whh_sw, pooled);
    // K4: classifier
    classify<<<1, 192, 0, stream>>>(pooled, W_cls, b_cls, out);
}
